// VisionMamba_74397423501373
// MI455X (gfx1250) — compile-verified, matrix operand order checked by reading, not run on hardware
//
#include <hip/hip_runtime.h>
#include <hip/hip_bf16.h>
#include <math.h>

typedef __attribute__((ext_vector_type(16))) __bf16 v16bf;
typedef __attribute__((ext_vector_type(8)))  float  v8f;

#define BB      8
#define LL      1024
#define MROWS   (BB * LL)        // 8192
#define DMODEL  384
#define DINNER  768
#define DSTATE  16
#define DTRANK  24
#define NDBL    (DTRANK + 2 * DSTATE)  // 56
#define EPSV    1e-5f

__device__ __forceinline__ unsigned short f2bf(float f) {
  union { float f; unsigned int u; } c; c.f = f;
  unsigned int u = c.u;
  return (unsigned short)((u + 0x7FFFu + ((u >> 16) & 1u)) >> 16);
}
__device__ __forceinline__ unsigned int pk2(float a, float b) {
  return (unsigned int)f2bf(a) | ((unsigned int)f2bf(b) << 16);
}
__device__ __forceinline__ float sigmoidf_(float x) { return 1.0f / (1.0f + __expf(-x)); }
__device__ __forceinline__ float softplusf_(float x) { return (x > 20.f) ? x : log1pf(__expf(x)); }
__device__ __forceinline__ float fget(const float4& v, int c) {
  switch (c & 3) { case 0: return v.x; case 1: return v.y; case 2: return v.z; default: return v.w; }
}

__device__ __forceinline__ void wait_asynccnt0() {
#if __has_builtin(__builtin_amdgcn_s_wait_asynccnt)
  __builtin_amdgcn_s_wait_asynccnt(0);
#else
  asm volatile("s_wait_asynccnt 0x0" ::: "memory");
#endif
}

// one b128 async copy: global (16B) -> LDS (16B), tracked by ASYNCcnt
__device__ __forceinline__ void async_cp16(unsigned int lds_off, const unsigned short* g) {
  asm volatile("global_load_async_to_lds_b128 %0, %1, off"
               :: "v"(lds_off), "v"((unsigned long long)g) : "memory");
}

// ---------------------------------------------------------------------------
// f32 -> bf16 bulk convert (n % 4 == 0 at all call sites)
// ---------------------------------------------------------------------------
__global__ __launch_bounds__(256) void cvt_bf16_k(const float* __restrict__ src,
                                                  unsigned short* __restrict__ dst, int n) {
  const int i = (blockIdx.x * 256 + threadIdx.x) * 4;
  if (i < n) {
    const float4 f = *(const float4*)(src + i);
    *(uint2*)(dst + i) = make_uint2(pk2(f.x, f.y), pk2(f.z, f.w));
  }
}

// ---------------------------------------------------------------------------
// dst = bf16(a + b), elementwise (combine forward/backward scan outputs)
// ---------------------------------------------------------------------------
__global__ __launch_bounds__(256) void combine_bf16_k(const float* __restrict__ a,
                                                      const float* __restrict__ b,
                                                      unsigned short* __restrict__ dst, int n) {
  const int i = (blockIdx.x * 256 + threadIdx.x) * 4;
  if (i < n) {
    const float4 va = *(const float4*)(a + i);
    const float4 vb = *(const float4*)(b + i);
    *(uint2*)(dst + i) = make_uint2(pk2(va.x + vb.x, va.y + vb.y),
                                    pk2(va.z + vb.z, va.w + vb.w));
  }
}

// ---------------------------------------------------------------------------
// GEMM: C[M,N] = A[M,K] (bf16, lda) * W[N,K]^T (bf16) (+bias)(+softplus)
// Block tile 128x128, K step 32, 8 waves, 8 WMMA/wave/step, double-buffered
// LDS fed by GLOBAL_LOAD_ASYNC_TO_LDS_B128. Tail guards are chunk-granular
// (8 bf16 = 16B): valid chunk -> async copy, invalid -> zero ds_store_b128.
// B fragments are software-pipelined through 2 register sets so ds_load
// latency overlaps WMMA. Optional bf16 copy of C.
// ---------------------------------------------------------------------------
template <bool TAIL>
__global__ __launch_bounds__(256) void gemm_wmma_bf16(
    const unsigned short* __restrict__ A, int lda,
    const unsigned short* __restrict__ W,
    const float* __restrict__ bias,
    float* __restrict__ C, unsigned short* __restrict__ Cbf,
    int M, int N, int K, int act)
{
  __shared__ __align__(16) unsigned short As[2][128][40];  // 32 + 8 pad
  __shared__ __align__(16) unsigned short Bs[2][128][40];

  const int tid  = threadIdx.x;
  const int wave = tid >> 5;
  const int lane = tid & 31;
  const int m0 = blockIdx.y * 128;
  const int n0 = blockIdx.x * 128;

  const int srow  = tid >> 1;          // 0..127
  const int scol0 = (tid & 1) << 4;    // 0 or 16
  const int kTiles = (K + 31) >> 5;

  auto stage = [&](int kt, int buf) {
    const int k0 = kt << 5;
    #pragma unroll
    for (int j = 0; j < 2; ++j) {
      const int col = scol0 + j * 8;
      // A chunk (rows always valid; K chunk-granular)
      const unsigned int la = (unsigned int)(unsigned long long)&As[buf][srow][col];
      if (!TAIL || (k0 + col + 8 <= K))
        async_cp16(la, A + (size_t)(m0 + srow) * lda + k0 + col);
      else
        *(uint4*)&As[buf][srow][col] = make_uint4(0u, 0u, 0u, 0u);
      // B chunk (row + K guards)
      const unsigned int lb = (unsigned int)(unsigned long long)&Bs[buf][srow][col];
      if (!TAIL || ((n0 + srow < N) && (k0 + col + 8 <= K)))
        async_cp16(lb, W + (size_t)(n0 + srow) * K + k0 + col);
      else
        *(uint4*)&Bs[buf][srow][col] = make_uint4(0u, 0u, 0u, 0u);
    }
  };

  v8f acc[8];
  #pragma unroll
  for (int i = 0; i < 8; ++i)
    #pragma unroll
    for (int j = 0; j < 8; ++j) acc[i][j] = 0.0f;

  const int fr = lane & 15;            // A row / B col within 16
  const int kh = (lane >> 4) << 3;     // K half: 0 or 8

  stage(0, 0);
  wait_asynccnt0();
  __syncthreads();

  union U { uint4 q[2]; v16bf v; };
  for (int kt = 0; kt < kTiles; ++kt) {
    const int buf = kt & 1;
    if (kt + 1 < kTiles) stage(kt + 1, buf ^ 1);   // async prefetch next tile

    U ua;
    ua.q[0] = *(const uint4*)&As[buf][wave * 16 + fr][kh];
    ua.q[1] = *(const uint4*)&As[buf][wave * 16 + fr][kh + 16];
    U ub[2];
    ub[0].q[0] = *(const uint4*)&Bs[buf][fr][kh];
    ub[0].q[1] = *(const uint4*)&Bs[buf][fr][kh + 16];
    #pragma unroll
    for (int nt = 0; nt < 8; ++nt) {
      if (nt < 7) {   // preload next B fragment into the other register set
        ub[(nt + 1) & 1].q[0] = *(const uint4*)&Bs[buf][(nt + 1) * 16 + fr][kh];
        ub[(nt + 1) & 1].q[1] = *(const uint4*)&Bs[buf][(nt + 1) * 16 + fr][kh + 16];
      }
      acc[nt] = __builtin_amdgcn_wmma_f32_16x16x32_bf16(
          false, ua.v, false, ub[nt & 1].v, (short)0, acc[nt], false, false);
    }
    wait_asynccnt0();                  // own async copies done
    __syncthreads();                   // all waves staged + done reading
  }

  // store: VGPR r -> M = r (lanes<16) / r+8 (lanes>=16); N = lane&15
  const int rbase = m0 + wave * 16 + ((lane >> 4) << 3);
  #pragma unroll
  for (int nt = 0; nt < 8; ++nt) {
    const int col = n0 + nt * 16 + fr;
    if (!TAIL || col < N) {
      const float bv = bias ? bias[col] : 0.0f;
      #pragma unroll
      for (int r = 0; r < 8; ++r) {
        float v = acc[nt][r] + bv;
        if (act == 1) v = softplusf_(v);
        const size_t idx = (size_t)(rbase + r) * N + col;
        C[idx] = v;
        if (Cbf) Cbf[idx] = f2bf(v);
      }
    }
  }
}

// ---------------------------------------------------------------------------
// Residual add + RMSNorm over D_MODEL=384. One block (128 thr) per row.
// ---------------------------------------------------------------------------
__global__ __launch_bounds__(128) void resnorm_k(
    float* __restrict__ R, const float* __restrict__ H,
    const float* __restrict__ w, float* __restrict__ out,
    unsigned short* __restrict__ out_bf, int addH)
{
  __shared__ float red[128];
  const int row = blockIdx.x;
  const int t = threadIdx.x;
  float vals[3];
  float ss = 0.f;
  #pragma unroll
  for (int i = 0; i < 3; ++i) {
    const int c = t + i * 128;
    float r = R[(size_t)row * DMODEL + c];
    if (addH) { r += H[(size_t)row * DMODEL + c]; R[(size_t)row * DMODEL + c] = r; }
    vals[i] = r;
    ss += r * r;
  }
  red[t] = ss;
  __syncthreads();
  for (int s = 64; s > 0; s >>= 1) {
    if (t < s) red[t] += red[t + s];
    __syncthreads();
  }
  const float scale = rsqrtf(red[0] / (float)DMODEL + EPSV);
  #pragma unroll
  for (int i = 0; i < 3; ++i) {
    const int c = t + i * 128;
    const float v = vals[i] * scale * w[c];
    if (out)    out[(size_t)row * DMODEL + c] = v;
    if (out_bf) out_bf[(size_t)row * DMODEL + c] = f2bf(v);
  }
}

// ---------------------------------------------------------------------------
// Depthwise causal conv (K=4) + SiLU; dir=0 taps l-3+j, dir=1 taps l+3-j.
// ---------------------------------------------------------------------------
__global__ __launch_bounds__(256) void conv_silu_k(
    const float* __restrict__ xz, const float* __restrict__ w,
    const float* __restrict__ cb, float* __restrict__ xc,
    unsigned short* __restrict__ xc_bf, int dir)
{
  const int idx = blockIdx.x * 256 + threadIdx.x;
  if (idx >= MROWS * DINNER) return;
  const int d  = idx % DINNER;
  const int bl = idx / DINNER;
  const int b  = bl / LL;
  const int l  = bl % LL;
  float acc = cb[d];
  #pragma unroll
  for (int j = 0; j < 4; ++j) {
    const int src = dir ? (l + 3 - j) : (l - 3 + j);
    if (src >= 0 && src < LL)
      acc += w[d * 4 + j] * xz[(size_t)(b * LL + src) * (2 * DINNER) + d];
  }
  const float v = acc * sigmoidf_(acc);
  xc[(size_t)bl * DINNER + d] = v;
  xc_bf[(size_t)bl * DINNER + d] = f2bf(v);
}

// ---------------------------------------------------------------------------
// Selective scan + SiLU(z) gate, both directions in one launch (blockIdx.z).
// Barrier-free: B_t/C_t are 8 aligned float4 broadcast loads (L2-resident),
// double-buffered across a 2-unrolled time loop to hide load latency.
// dir=1 iterates p=L-1..0 and writes at p (implicit un-flip).
// ---------------------------------------------------------------------------
__global__ __launch_bounds__(256) void scan_gate2_k(
    const float* __restrict__ xcF, const float* __restrict__ xcB,
    const float* __restrict__ dF,  const float* __restrict__ dB,
    const float* __restrict__ dblF,const float* __restrict__ dblB,
    const float* __restrict__ alF, const float* __restrict__ alB,
    const float* __restrict__ DF,  const float* __restrict__ DB,
    const float* __restrict__ xz,
    float* __restrict__ YF, float* __restrict__ YB)
{
  const int dir = blockIdx.z;
  const float* xc    = dir ? xcB  : xcF;
  const float* delta = dir ? dB   : dF;
  const float* dbl   = dir ? dblB : dblF;
  const float* alog  = dir ? alB  : alF;
  const float* Dp    = dir ? DB   : DF;
  float*       Y     = dir ? YB   : YF;

  const int b = blockIdx.y;
  const int d = blockIdx.x * 256 + threadIdx.x;

  float Avals[DSTATE];
  #pragma unroll
  for (int n = 0; n < DSTATE; ++n) Avals[n] = -__expf(alog[(size_t)d * DSTATE + n]);
  const float Dd = Dp[d];
  float h[DSTATE];
  #pragma unroll
  for (int n = 0; n < DSTATE; ++n) h[n] = 0.f;

  auto ldrow = [&](int p, float4* q) {
    const float4* bc = (const float4*)(dbl + ((size_t)b * LL + p) * NDBL + DTRANK);
    #pragma unroll
    for (int i = 0; i < 8; ++i) q[i] = bc[i];   // q[0..3]=B, q[4..7]=C
  };
  auto step = [&](int p, const float4* q) {
    const size_t row = (size_t)b * LL + p;
    const float dt = delta[row * DINNER + d];
    const float u  = xc[row * DINNER + d];
    const float du = dt * u;
    float y = 0.f;
    #pragma unroll
    for (int n = 0; n < DSTATE; ++n) {
      h[n] = h[n] * __expf(dt * Avals[n]) + du * fget(q[n >> 2], n);
      y += h[n] * fget(q[4 + (n >> 2)], n);
    }
    y += u * Dd;
    const float z = xz[row * (2 * DINNER) + DINNER + d];
    Y[row * DINNER + d] = y * (z * sigmoidf_(z));
  };

  float4 q0[8], q1[8];
  ldrow(dir ? LL - 1 : 0, q0);
  for (int t = 0; t < LL; t += 2) {
    const int pA = dir ? (LL - 1 - t) : t;
    const int pB = dir ? (LL - 2 - t) : t + 1;
    ldrow(pB, q1);                 // prefetch step t+1 while computing t
    step(pA, q0);
    if (t + 2 < LL) ldrow(dir ? (LL - 3 - t) : t + 2, q0);
    step(pB, q1);
  }
}

// ---------------------------------------------------------------------------
extern "C" void kernel_launch(void* const* d_in, const int* in_sizes, int n_in,
                              void* d_out, int out_size, void* d_ws, size_t ws_size,
                              hipStream_t stream)
{
  const float* x         = (const float*)d_in[0];
  const float* embed_w   = (const float*)d_in[1];
  const float* embed_b   = (const float*)d_in[2];
  const float* in_proj_w = (const float*)d_in[3];
  const float* conv_w    = (const float*)d_in[4];
  const float* conv_b    = (const float*)d_in[5];
  const float* x_proj_w  = (const float*)d_in[6];
  const float* dt_proj_w = (const float*)d_in[7];
  const float* dt_proj_b = (const float*)d_in[8];
  const float* A_log     = (const float*)d_in[9];
  const float* Dvec      = (const float*)d_in[10];
  const float* conv_w_b  = (const float*)d_in[11];
  const float* conv_b_b  = (const float*)d_in[12];
  const float* x_proj_wb = (const float*)d_in[13];
  const float* dt_proj_wb= (const float*)d_in[14];
  const float* dt_proj_bb= (const float*)d_in[15];
  const float* A_log_b   = (const float*)d_in[16];
  const float* D_bvec    = (const float*)d_in[17];
  const float* out_proj_w= (const float*)d_in[18];
  const float* norm_w    = (const float*)d_in[19];
  const float* normf_w   = (const float*)d_in[20];
  float* out = (float*)d_out;

  // ---- workspace: f32 region then bf16 (ushort) region, all 16B aligned ----
  float* ws = (float*)d_ws;
  size_t off = 0;
  float* R      = ws + off; off += (size_t)MROWS * DMODEL;
  float* XZ     = ws + off; off += (size_t)MROWS * 2 * DINNER;
  float* XCf    = ws + off; off += (size_t)MROWS * DINNER;
  float* XCb    = ws + off; off += (size_t)MROWS * DINNER;
  float* DBLf   = ws + off; off += (size_t)MROWS * NDBL;
  float* DBLb   = ws + off; off += (size_t)MROWS * NDBL;
  float* DELTAf = ws + off; off += (size_t)MROWS * DINNER;
  float* DELTAb = ws + off; off += (size_t)MROWS * DINNER;
  float* YF     = ws + off; off += (size_t)MROWS * DINNER;
  float* YB     = ws + off; off += (size_t)MROWS * DINNER;
  float* Hb     = ws + off; off += (size_t)MROWS * DMODEL;

  unsigned short* us = (unsigned short*)(ws + off);
  size_t uo = 0;
  unsigned short* XBF    = us + uo; uo += (size_t)MROWS * 4096;
  unsigned short* EW     = us + uo; uo += (size_t)DMODEL * 4096;
  unsigned short* IPW    = us + uo; uo += (size_t)8 * 2 * DINNER * DMODEL;
  unsigned short* XPWf   = us + uo; uo += (size_t)8 * NDBL * DINNER;
  unsigned short* XPWb   = us + uo; uo += (size_t)8 * NDBL * DINNER;
  unsigned short* DTWf   = us + uo; uo += (size_t)8 * DINNER * DTRANK;
  unsigned short* DTWb   = us + uo; uo += (size_t)8 * DINNER * DTRANK;
  unsigned short* OPW    = us + uo; uo += (size_t)8 * DMODEL * DINNER;
  unsigned short* HNbf   = us + uo; uo += (size_t)MROWS * DMODEL;
  unsigned short* XCfbf  = us + uo; uo += (size_t)MROWS * DINNER;
  unsigned short* XCbbf  = us + uo; uo += (size_t)MROWS * DINNER;
  unsigned short* DBLfbf = us + uo; uo += (size_t)MROWS * NDBL;
  unsigned short* DBLbbf = us + uo; uo += (size_t)MROWS * NDBL;
  unsigned short* YSbf   = us + uo; uo += (size_t)MROWS * DINNER;

  auto cvt = [&](const float* s, unsigned short* d, size_t n) {
    cvt_bf16_k<<<(unsigned)((n / 4 + 255) / 256), 256, 0, stream>>>(s, d, (int)n);
  };
  auto gemm = [&](const unsigned short* A, int lda, const unsigned short* W,
                  const float* bias, float* C, unsigned short* Cbf,
                  int M, int N, int K, int act, bool tail) {
    dim3 grid((N + 127) / 128, M / 128);
    if (tail)
      gemm_wmma_bf16<true ><<<grid, 256, 0, stream>>>(A, lda, W, bias, C, Cbf, M, N, K, act);
    else
      gemm_wmma_bf16<false><<<grid, 256, 0, stream>>>(A, lda, W, bias, C, Cbf, M, N, K, act);
  };

  // ---- per-launch bf16 conversions (deterministic, no caching) ----
  cvt(x,          XBF,  (size_t)MROWS * 4096);
  cvt(embed_w,    EW,   (size_t)DMODEL * 4096);
  cvt(in_proj_w,  IPW,  (size_t)8 * 2 * DINNER * DMODEL);
  cvt(x_proj_w,   XPWf, (size_t)8 * NDBL * DINNER);
  cvt(x_proj_wb,  XPWb, (size_t)8 * NDBL * DINNER);
  cvt(dt_proj_w,  DTWf, (size_t)8 * DINNER * DTRANK);
  cvt(dt_proj_wb, DTWb, (size_t)8 * DINNER * DTRANK);
  cvt(out_proj_w, OPW,  (size_t)8 * DMODEL * DINNER);

  // Embedding: R = x @ embed_w^T + embed_b  (residual starts as h)
  gemm(XBF, 4096, EW, embed_b, R, nullptr, MROWS, DMODEL, 4096, 0, false);

  const int convThreads = MROWS * DINNER;
  for (int i = 0; i < 8; ++i) {
    resnorm_k<<<MROWS, 128, 0, stream>>>(R, Hb, norm_w + (size_t)i * DMODEL,
                                         nullptr, HNbf, i > 0 ? 1 : 0);
    gemm(HNbf, DMODEL, IPW + (size_t)i * 2 * DINNER * DMODEL, nullptr,
         XZ, nullptr, MROWS, 2 * DINNER, DMODEL, 0, false);

    // both directions' pointwise pipelines, then one fused dual-direction scan
    conv_silu_k<<<(convThreads + 255) / 256, 256, 0, stream>>>(
        XZ, conv_w   + (size_t)i * DINNER * 4, conv_b   + (size_t)i * DINNER, XCf, XCfbf, 0);
    conv_silu_k<<<(convThreads + 255) / 256, 256, 0, stream>>>(
        XZ, conv_w_b + (size_t)i * DINNER * 4, conv_b_b + (size_t)i * DINNER, XCb, XCbbf, 1);

    gemm(XCfbf, DINNER, XPWf + (size_t)i * NDBL * DINNER, nullptr,
         DBLf, DBLfbf, MROWS, NDBL, DINNER, 0, true);
    gemm(XCbbf, DINNER, XPWb + (size_t)i * NDBL * DINNER, nullptr,
         DBLb, DBLbbf, MROWS, NDBL, DINNER, 0, true);

    gemm(DBLfbf, NDBL, DTWf + (size_t)i * DINNER * DTRANK, dt_proj_b  + (size_t)i * DINNER,
         DELTAf, nullptr, MROWS, DINNER, DTRANK, 1, true);
    gemm(DBLbbf, NDBL, DTWb + (size_t)i * DINNER * DTRANK, dt_proj_bb + (size_t)i * DINNER,
         DELTAb, nullptr, MROWS, DINNER, DTRANK, 1, true);

    scan_gate2_k<<<dim3(DINNER / 256, BB, 2), 256, 0, stream>>>(
        XCf, XCb, DELTAf, DELTAb, DBLf, DBLb,
        A_log   + (size_t)i * DINNER * DSTATE, A_log_b + (size_t)i * DINNER * DSTATE,
        Dvec    + (size_t)i * DINNER,          D_bvec  + (size_t)i * DINNER,
        XZ, YF, YB);

    combine_bf16_k<<<(convThreads / 4 + 255) / 256, 256, 0, stream>>>(
        YF, YB, YSbf, convThreads);

    gemm(YSbf, DINNER, OPW + (size_t)i * DMODEL * DINNER, nullptr,
         Hb, nullptr, MROWS, DMODEL, DINNER, 0, false);
  }
  resnorm_k<<<MROWS, 128, 0, stream>>>(R, Hb, normf_w, out, nullptr, 1);
}
